// GroupedQueryAttention_86620900426254
// MI455X (gfx1250) — matrix operand order repaired and checked by reading, not compiled, not run on hardware
//
#include <hip/hip_runtime.h>

// ---------------------------------------------------------------------------
// GQA forward for MI455X (gfx1250, wave32, WMMA + TDM, double-buffered LDS).
// bf16 compute, fp32 accumulate.
// ---------------------------------------------------------------------------

#define DMODEL 2048
#define NH 16
#define NKV 4
#define HD 128
#define SEQ 2048
#define BATCH 2
#define MTOT (BATCH * SEQ) // 4096

typedef __attribute__((ext_vector_type(16))) __bf16 bf16x16;
typedef __attribute__((ext_vector_type(8))) float f32x8;

#if defined(__gfx1250__) && __has_builtin(__builtin_amdgcn_tensor_load_to_lds)
#define USE_TDM 1
#endif

#if defined(USE_TDM)
typedef __attribute__((ext_vector_type(4))) unsigned int tdm_u32x4;
typedef __attribute__((ext_vector_type(8))) int tdm_i32x8;
typedef __attribute__((ext_vector_type(4))) int tdm_i32x4;

// Issue a 2D tile load via the Tensor Data Mover (D# packing per CDNA5 ISA
// ch.8: group0 = {count/lds_addr/global_addr/type=2}, group1 = dims/strides/pad).
// All args must be wave-uniform. Tracked by TENSORcnt.
__device__ inline void tdm_load_2d(const void* gptr, void* lptr,
                                   unsigned tensor_d0, unsigned tensor_d1,
                                   unsigned tile_d0, unsigned tile_d1,
                                   unsigned stride0,         // elements between rows
                                   unsigned pad_en,
                                   unsigned pad_interval,    // 0=2dw,1=4,2=8,3=16,...,7=256
                                   unsigned pad_amount) {    // dwords-1
    unsigned long long ga = (unsigned long long)gptr;
    unsigned lds = (unsigned)(unsigned long long)lptr; // low 32 bits = LDS offset

    tdm_u32x4 g0;
    g0[0] = 1u;                                   // count=1, user mode, no gather
    g0[1] = lds;                                  // lds_addr [63:32]
    g0[2] = (unsigned)(ga & 0xFFFFFFFFull);       // global_addr[31:0]
    g0[3] = (unsigned)((ga >> 32) & 0x01FFFFFFull) | (2u << 30); // addr[56:32] | type=2

    const unsigned flags = (1u << 16)             // data_size = 1 -> 2 bytes (bf16)
                         | (pad_en << 20)
                         | (pad_interval << 22)
                         | (pad_amount << 25);    // workgroup_mask = 0 (no cluster)
    unsigned long long q0 = (unsigned long long)flags
                          | ((unsigned long long)(tensor_d0 & 0xFFFFu) << 48);
    unsigned long long q1 = (unsigned long long)(tensor_d0 >> 16)
                          | ((unsigned long long)tensor_d1 << 16)
                          | ((unsigned long long)(tile_d0 & 0xFFFFu) << 48);
    unsigned long long q2 = (unsigned long long)(tile_d1 & 0xFFFFu)   // tile_dim2 = 0
                          | ((unsigned long long)stride0 << 32);      // stride0[31:0]
    unsigned long long q3 = 0ull;                 // stride0 hi = 0, dim1_stride = 0

    tdm_i32x8 g1;
    g1[0] = (int)(q0 & 0xFFFFFFFFull); g1[1] = (int)(q0 >> 32);
    g1[2] = (int)(q1 & 0xFFFFFFFFull); g1[3] = (int)(q1 >> 32);
    g1[4] = (int)(q2 & 0xFFFFFFFFull); g1[5] = (int)(q2 >> 32);
    g1[6] = (int)(q3 & 0xFFFFFFFFull); g1[7] = (int)(q3 >> 32);

    tdm_i32x4 z4; z4[0] = z4[1] = z4[2] = z4[3] = 0;
#if __has_include(<hip/amd_detail/amd_gfx1250_TDM.h>)
    // therock-10.0 headers -> 6-arg builtin
    tdm_i32x8 z8;
    for (int i = 0; i < 8; ++i) z8[i] = 0;
    __builtin_amdgcn_tensor_load_to_lds(g0, g1, z4, z4, z8, 0);
#else
    // ROCm 7.2 clang-22 -> 5-arg builtin
    __builtin_amdgcn_tensor_load_to_lds(g0, g1, z4, z4, 0);
#endif
}

__device__ inline void tdm_wait() {
#if __has_builtin(__builtin_amdgcn_s_wait_tensorcnt)
    __builtin_amdgcn_s_wait_tensorcnt(0);
#endif
}
#endif // USE_TDM

union BFrag {
    bf16x16 v;
    unsigned int u[8];
    __bf16 h[16];
};

__device__ inline f32x8 zero8() {
    f32x8 z;
#pragma unroll
    for (int i = 0; i < 8; ++i) z[i] = 0.0f;
    return z;
}

__device__ inline f32x8 wmma_bf16(bf16x16 a, bf16x16 b, f32x8 c) {
    // v_wmma_f32_16x16x32_bf16  D = A*B + C   (wave32)
    return __builtin_amdgcn_wmma_f32_16x16x32_bf16(
        /*neg_a=*/false, a, /*neg_b=*/false, b,
        /*c_mod=*/(short)0, c, /*reuse_a=*/false, /*reuse_b=*/false);
}

// A-matrix 16x32 bf16 fragment, from row-major [16][>=32] with element stride.
__device__ inline bf16x16 load_a_frag(const __bf16* base, int stride, int lane) {
    const int row = lane & 15;
    const int kh8 = (lane >> 4) << 3;
    const __bf16* p = base + row * stride;
    BFrag f;
#pragma unroll
    for (int i = 0; i < 4; ++i)
        f.u[i] = *(const unsigned int*)(p + kh8 + 2 * i);
#pragma unroll
    for (int i = 0; i < 4; ++i)
        f.u[4 + i] = *(const unsigned int*)(p + 16 + kh8 + 2 * i);
    return f.v;
}

// B-matrix 32x16 bf16 fragment from K-contiguous (transposed) storage Bt[n][k].
__device__ inline bf16x16 load_b_frag_t(const __bf16* base, int stride, int lane) {
    const int col = lane & 15;
    const int kh8 = (lane >> 4) << 3;
    const __bf16* p = base + col * stride;
    BFrag f;
#pragma unroll
    for (int i = 0; i < 4; ++i)
        f.u[i] = *(const unsigned int*)(p + kh8 + 2 * i);
#pragma unroll
    for (int i = 0; i < 4; ++i)
        f.u[4 + i] = *(const unsigned int*)(p + 16 + kh8 + 2 * i);
    return f.v;
}

// ---------------------------------------------------------------------------
// Kernel 0: fp32 -> bf16 conversion (n divisible by 4)
// ---------------------------------------------------------------------------
__global__ __launch_bounds__(256) void cvt_f32_bf16(const float* __restrict__ s,
                                                    __bf16* __restrict__ d, int n) {
    int i = (blockIdx.x * 256 + threadIdx.x) * 4;
    if (i < n) {
        float4 v = *(const float4*)(s + i);
        d[i + 0] = (__bf16)v.x;
        d[i + 1] = (__bf16)v.y;
        d[i + 2] = (__bf16)v.z;
        d[i + 3] = (__bf16)v.w;
    }
}

// ---------------------------------------------------------------------------
// Kernel 1: fused QKV projection + RMSNorm + RoPE.  Double-buffered pipeline:
// issue TDM/A + staged B for tile i+1, run WMMAs on tile i, then wait+barrier.
// grid = (MTOT/64, 24): y in [0,16) -> Q head, [16,20) -> K head, [20,24) -> V head
// ---------------------------------------------------------------------------
__global__ __launch_bounds__(256) void qkv_kernel(
    const __bf16* __restrict__ xb, const __bf16* __restrict__ wqb,
    const __bf16* __restrict__ wkb, const __bf16* __restrict__ wvb,
    const float* __restrict__ qw, const float* __restrict__ kw,
    __bf16* __restrict__ qout, __bf16* __restrict__ kout, __bf16* __restrict__ vout) {
    __shared__ __bf16 As[2][64][40];   // 64x32 tiles, rows 64B data + 16B pad
    __shared__ __bf16 Bt[2][128][40];  // transposed 32x128 tiles, padded
    __shared__ float Cs[64][132];      // fp32 result tile for epilogue

    const int tid = threadIdx.x;
    const int w = tid >> 5;
    const int lane = tid & 31;
    const int mb = blockIdx.x * 64;
    const int hy = blockIdx.y;

    const __bf16* W;
    int ldw, ncol0;
    if (hy < 16) { W = wqb; ldw = NH * HD;  ncol0 = hy * HD; }
    else if (hy < 20) { W = wkb; ldw = NKV * HD; ncol0 = (hy - 16) * HD; }
    else { W = wvb; ldw = NKV * HD; ncol0 = (hy - 20) * HD; }

    const int m0 = (w >> 1) << 4;
    const int n0 = (w & 1) << 6;

    auto stage_a = [&](int kk, int buf) {
#if defined(USE_TDM)
        if (w == 0)
            tdm_load_2d(xb + (size_t)mb * DMODEL + kk, &As[buf][0][0],
                        /*tensor_d0=*/DMODEL, /*tensor_d1=*/MTOT,
                        /*tile_d0=*/32, /*tile_d1=*/64, /*stride0=*/DMODEL,
                        /*pad_en=*/1, /*pad_interval=*/3 /*16dw*/, /*pad_amount=*/3 /*4dw*/);
#else
        const int row = tid >> 2, cg = (tid & 3) << 3;
        uint4 d = *(const uint4*)(xb + (size_t)(mb + row) * DMODEL + kk + cg);
        unsigned int* dst = (unsigned int*)(&As[buf][row][cg]);
        dst[0] = d.x; dst[1] = d.y; dst[2] = d.z; dst[3] = d.w;
#endif
    };
    auto stage_b = [&](int kk, int buf) {
        const int r = tid & 31, cg = (tid >> 5) << 4;
        const __bf16* src = W + (size_t)(kk + r) * ldw + ncol0 + cg;
        BFrag u;
        *(uint4*)&u.u[0] = *(const uint4*)(src);
        *(uint4*)&u.u[4] = *(const uint4*)(src + 8);
#pragma unroll
        for (int c = 0; c < 16; ++c) Bt[buf][cg + c][r] = u.h[c];
        __builtin_prefetch(src + (size_t)32 * ldw, 0, 1); // next-next tile
    };

    f32x8 acc[4];
#pragma unroll
    for (int t = 0; t < 4; ++t) acc[t] = zero8();

    // prologue: tile 0 into buffer 0
    stage_a(0, 0);
    stage_b(0, 0);
#if defined(USE_TDM)
    tdm_wait();
#endif
    __syncthreads();

    for (int kk = 0; kk < DMODEL; kk += 32) {
        const int buf = (kk >> 5) & 1;
        if (kk + 32 < DMODEL) {           // stage tile i+1 into the other buffer
            stage_a(kk + 32, buf ^ 1);
            stage_b(kk + 32, buf ^ 1);
        }
        bf16x16 a = load_a_frag(&As[buf][m0][0], 40, lane);
#pragma unroll
        for (int t = 0; t < 4; ++t) {
            bf16x16 b = load_b_frag_t(&Bt[buf][n0 + t * 16][0], 40, lane);
            acc[t] = wmma_bf16(a, b, acc[t]);
        }
#if defined(USE_TDM)
        tdm_wait();                       // tile i+1 DMA complete
#endif
        __syncthreads();                  // publish + protect buffers
    }

    // spill fp32 accumulators to LDS (C layout: row = i + 8*(lane>=16), col = lane&15)
    {
        const int kh8 = (lane >> 4) << 3;
        const int cl = lane & 15;
#pragma unroll
        for (int t = 0; t < 4; ++t)
#pragma unroll
            for (int i = 0; i < 8; ++i)
                Cs[m0 + kh8 + i][n0 + t * 16 + cl] = acc[t][i];
    }
    __syncthreads();

    // epilogue: one thread per token row
    if (tid < 64) {
        const int gtok = mb + tid;
        const int b = gtok >> 11;     // / SEQ
        const int tpos = gtok & 2047; // % SEQ
        if (hy >= 20) {
            __bf16* dst = vout + ((size_t)(b * NKV + (hy - 20)) * SEQ + tpos) * HD;
            for (int j = 0; j < HD; ++j) dst[j] = (__bf16)Cs[tid][j];
        } else {
            const float* nw = (hy < 16) ? qw : kw;
            float ss = 0.0f;
            for (int j = 0; j < HD; ++j) { float v = Cs[tid][j]; ss += v * v; }
            const float rms = rsqrtf(ss * (1.0f / HD) + 1e-5f);
            __bf16* dst;
            if (hy < 16) dst = qout + ((size_t)(b * NH + hy) * SEQ + tpos) * HD;
            else         dst = kout + ((size_t)(b * NKV + (hy - 16)) * SEQ + tpos) * HD;
            for (int j = 0; j < 64; ++j) {
                const float invf = __powf(500000.0f, -((float)(2 * j)) * (1.0f / HD));
                float sn, cs;
                __sincosf((float)tpos * invf, &sn, &cs);
                const float x1 = Cs[tid][j] * rms * nw[j];
                const float x2 = Cs[tid][j + 64] * rms * nw[j + 64];
                dst[j]      = (__bf16)(x1 * cs - x2 * sn);
                dst[j + 64] = (__bf16)(x2 * cs + x1 * sn);
            }
        }
    }
}

// ---------------------------------------------------------------------------
// Kernel 2: causal GQA flash attention, double-buffered K/V staging.
// grid = (SEQ/64, NH, BATCH), block = 128 (4 waves). Each wave owns 16 q rows.
// ---------------------------------------------------------------------------
__global__ __launch_bounds__(128) void flash_kernel(
    const __bf16* __restrict__ qb, const __bf16* __restrict__ kb,
    const __bf16* __restrict__ vb, __bf16* __restrict__ ob) {
    __shared__ __bf16 Ks[2][64][132];   // [key][dim], rows 256B data + 8B pad
    __shared__ __bf16 Vt[2][128][68];   // [dim][key] (B-frag source for P@V)
    __shared__ __bf16 Ps[4][16][68];    // per-wave P tile in row-major

    const int tid = threadIdx.x;
    const int w = tid >> 5;
    const int lane = tid & 31;
    const int qt = blockIdx.x;
    const int h = blockIdx.y;
    const int b = blockIdx.z;
    const int qt0 = qt * 64;
    const int hkv = h >> 2; // NH/NKV = 4

    const __bf16* Qbase = qb + (size_t)(b * NH + h) * SEQ * HD;
    const __bf16* Kbase = kb + (size_t)(b * NKV + hkv) * SEQ * HD;
    const __bf16* Vbase = vb + (size_t)(b * NKV + hkv) * SEQ * HD;

    const int kh8 = (lane >> 4) << 3;
    const int cl = lane & 15;

    auto stage_kv = [&](int kt0, int buf) {
        const int key = tid >> 1, dg = (tid & 1) << 6;
#if defined(USE_TDM)
        // K tile (64x128) via Tensor Data Mover; pad 8B per 256B row.
        if (w == 0)
            tdm_load_2d(Kbase + (size_t)kt0 * HD, &Ks[buf][0][0],
                        /*tensor_d0=*/HD, /*tensor_d1=*/SEQ,
                        /*tile_d0=*/HD, /*tile_d1=*/64, /*stride0=*/HD,
                        /*pad_en=*/1, /*pad_interval=*/5 /*64dw*/, /*pad_amount=*/1 /*2dw*/);
#else
        const __bf16* ksrc = Kbase + (size_t)(kt0 + key) * HD + dg;
        unsigned int* kdst = (unsigned int*)(&Ks[buf][key][dg]);
#pragma unroll
        for (int i = 0; i < 8; ++i) {
            uint4 d = ((const uint4*)ksrc)[i];
            kdst[i * 4 + 0] = d.x; kdst[i * 4 + 1] = d.y;
            kdst[i * 4 + 2] = d.z; kdst[i * 4 + 3] = d.w;
        }
#endif
        const __bf16* vsrc = Vbase + (size_t)(kt0 + key) * HD + dg;
#pragma unroll
        for (int i = 0; i < 8; ++i) {
            BFrag u;
            *(uint4*)&u.u[0] = ((const uint4*)vsrc)[i];
#pragma unroll
            for (int e = 0; e < 8; ++e) Vt[buf][dg + i * 8 + e][key] = u.h[e];
        }
        __builtin_prefetch(vsrc + (size_t)64 * HD, 0, 1); // tile after next
    };

    // preload Q fragments (16 rows x 128 dim -> 4 A-frags)
    const int qrow = qt0 + w * 16 + cl;
    bf16x16 qf[4];
#pragma unroll
    for (int s = 0; s < 4; ++s) {
        const __bf16* p = Qbase + (size_t)qrow * HD + s * 32;
        BFrag f;
#pragma unroll
        for (int i = 0; i < 4; ++i) f.u[i] = *(const unsigned int*)(p + kh8 + 2 * i);
#pragma unroll
        for (int i = 0; i < 4; ++i) f.u[4 + i] = *(const unsigned int*)(p + 16 + kh8 + 2 * i);
        qf[s] = f.v;
    }

    f32x8 o[8];
#pragma unroll
    for (int n = 0; n < 8; ++n) o[n] = zero8();
    float mrun[8], lrun[8];
#pragma unroll
    for (int i = 0; i < 8; ++i) { mrun[i] = -1e30f; lrun[i] = 0.0f; }

    const float scale = 0.08838834764831845f; // 1/sqrt(128)

    // prologue: key tile 0 into buffer 0
    stage_kv(0, 0);
#if defined(USE_TDM)
    tdm_wait();
#endif
    __syncthreads();

    for (int jt = 0; jt <= qt; ++jt) {
        const int buf = jt & 1;
        if (jt < qt) stage_kv((jt + 1) * 64, buf ^ 1); // overlap DMA with compute

        // ---- S = Q @ K^T (16x64 per wave) ----
        f32x8 sacc[4];
#pragma unroll
        for (int t = 0; t < 4; ++t) sacc[t] = zero8();
#pragma unroll
        for (int s = 0; s < 4; ++s) {
#pragma unroll
            for (int t = 0; t < 4; ++t) {
                bf16x16 bfr = load_b_frag_t(&Ks[buf][t * 16][s * 32], 132, lane);
                sacc[t] = wmma_bf16(qf[s], bfr, sacc[t]);
            }
        }

        // ---- scale + causal mask (diagonal tile only) ----
        const bool diag = (jt == qt);
#pragma unroll
        for (int t = 0; t < 4; ++t)
#pragma unroll
            for (int i = 0; i < 8; ++i) {
                float v = sacc[t][i] * scale;
                if (diag) {
                    const int row = w * 16 + kh8 + i;
                    const int col = t * 16 + cl;
                    if (col > row) v = -1e30f;
                }
                sacc[t][i] = v;
            }

        // ---- online softmax (rows live across 16 lanes of a half-wave) ----
#pragma unroll
        for (int i = 0; i < 8; ++i) {
            float mx = sacc[0][i];
#pragma unroll
            for (int t = 1; t < 4; ++t) mx = fmaxf(mx, sacc[t][i]);
#pragma unroll
            for (int off = 1; off < 16; off <<= 1)
                mx = fmaxf(mx, __shfl_xor(mx, off, 32));
            const float mnew = fmaxf(mrun[i], mx);
            const float al = __expf(mrun[i] - mnew);
            float rs = 0.0f;
#pragma unroll
            for (int t = 0; t < 4; ++t) {
                float p = __expf(sacc[t][i] - mnew);
                sacc[t][i] = p;
                rs += p;
            }
#pragma unroll
            for (int off = 1; off < 16; off <<= 1)
                rs += __shfl_xor(rs, off, 32);
            lrun[i] = lrun[i] * al + rs;
            mrun[i] = mnew;
#pragma unroll
            for (int n = 0; n < 8; ++n) o[n][i] *= al;
        }

        // ---- P: C-layout -> A-frag layout via per-wave LDS bounce ----
#pragma unroll
        for (int t = 0; t < 4; ++t)
#pragma unroll
            for (int i = 0; i < 8; ++i)
                Ps[w][kh8 + i][t * 16 + cl] = (__bf16)sacc[t][i];
        // intra-wave RAW on LDS; compiler inserts ds waits.

        // ---- O += P @ V ----
#pragma unroll
        for (int s = 0; s < 2; ++s) {
            bf16x16 pa = load_a_frag(&Ps[w][0][s * 32], 68, lane);
#pragma unroll
            for (int n = 0; n < 8; ++n) {
                bf16x16 bfr = load_b_frag_t(&Vt[buf][n * 16][s * 32], 68, lane);
                o[n] = wmma_bf16(pa, bfr, o[n]);
            }
        }
#if defined(USE_TDM)
        tdm_wait();                       // next K tile DMA complete
#endif
        __syncthreads();
    }

    // ---- finalize: O /= l, store bf16 [B][T][NH*HD] ----
#pragma unroll
    for (int i = 0; i < 8; ++i) {
        const float inv = 1.0f / lrun[i];
        const int row = qt0 + w * 16 + kh8 + i;
        __bf16* dst = ob + ((size_t)b * SEQ + row) * (NH * HD) + h * HD;
#pragma unroll
        for (int n = 0; n < 8; ++n)
            dst[n * 16 + cl] = (__bf16)(o[n][i] * inv);
    }
}

// ---------------------------------------------------------------------------
// Kernel 3: output projection (4096x2048) @ (2048x2048) -> fp32 out.
// grid = (MTOT/64, DMODEL/128), block = 256. Same double-buffered pipeline.
// ---------------------------------------------------------------------------
__global__ __launch_bounds__(256) void oproj_kernel(
    const __bf16* __restrict__ ab, const __bf16* __restrict__ wob,
    float* __restrict__ out) {
    __shared__ __bf16 As[2][64][40];
    __shared__ __bf16 Bt[2][128][40];

    const int tid = threadIdx.x;
    const int w = tid >> 5;
    const int lane = tid & 31;
    const int mb = blockIdx.x * 64;
    const int ncol0 = blockIdx.y * 128;

    const int m0 = (w >> 1) << 4;
    const int n0 = (w & 1) << 6;

    auto stage_a = [&](int kk, int buf) {
#if defined(USE_TDM)
        if (w == 0)
            tdm_load_2d(ab + (size_t)mb * DMODEL + kk, &As[buf][0][0],
                        DMODEL, MTOT, 32, 64, DMODEL,
                        1, 3 /*16dw*/, 3 /*4dw*/);
#else
        const int row = tid >> 2, cg = (tid & 3) << 3;
        uint4 d = *(const uint4*)(ab + (size_t)(mb + row) * DMODEL + kk + cg);
        unsigned int* dst = (unsigned int*)(&As[buf][row][cg]);
        dst[0] = d.x; dst[1] = d.y; dst[2] = d.z; dst[3] = d.w;
#endif
    };
    auto stage_b = [&](int kk, int buf) {
        const int r = tid & 31, cg = (tid >> 5) << 4;
        const __bf16* src = wob + (size_t)(kk + r) * DMODEL + ncol0 + cg;
        BFrag u;
        *(uint4*)&u.u[0] = *(const uint4*)(src);
        *(uint4*)&u.u[4] = *(const uint4*)(src + 8);
#pragma unroll
        for (int c = 0; c < 16; ++c) Bt[buf][cg + c][r] = u.h[c];
        __builtin_prefetch(src + (size_t)32 * DMODEL, 0, 1);
    };

    f32x8 acc[4];
#pragma unroll
    for (int t = 0; t < 4; ++t) acc[t] = zero8();

    stage_a(0, 0);
    stage_b(0, 0);
#if defined(USE_TDM)
    tdm_wait();
#endif
    __syncthreads();

    for (int kk = 0; kk < DMODEL; kk += 32) {
        const int buf = (kk >> 5) & 1;
        if (kk + 32 < DMODEL) {
            stage_a(kk + 32, buf ^ 1);
            stage_b(kk + 32, buf ^ 1);
        }
        bf16x16 a = load_a_frag(&As[buf][m0][0], 40, lane);
#pragma unroll
        for (int t = 0; t < 4; ++t) {
            bf16x16 b = load_b_frag_t(&Bt[buf][n0 + t * 16][0], 40, lane);
            acc[t] = wmma_bf16(a, b, acc[t]);
        }
#if defined(USE_TDM)
        tdm_wait();
#endif
        __syncthreads();
    }

    const int kh8 = (lane >> 4) << 3;
    const int cl = lane & 15;
#pragma unroll
    for (int t = 0; t < 4; ++t)
#pragma unroll
        for (int i = 0; i < 8; ++i)
            out[(size_t)(mb + m0 + kh8 + i) * DMODEL + ncol0 + n0 + t * 16 + cl] =
                acc[t][i];
}

// ---------------------------------------------------------------------------
// Host-side launcher
// ---------------------------------------------------------------------------
extern "C" void kernel_launch(void* const* d_in, const int* in_sizes, int n_in,
                              void* d_out, int out_size, void* d_ws, size_t ws_size,
                              hipStream_t stream) {
    const float* x  = (const float*)d_in[0];
    const float* Wq = (const float*)d_in[1];
    const float* Wk = (const float*)d_in[2];
    const float* Wv = (const float*)d_in[3];
    const float* Wo = (const float*)d_in[4];
    const float* qw = (const float*)d_in[5];
    const float* kw = (const float*)d_in[6];
    float* out = (float*)d_out;

    char* ws = (char*)d_ws;
    size_t off = 0;
    auto alloc = [&](size_t bytes) -> void* {
        void* p = ws + off;
        off += (bytes + 255) & ~(size_t)255;
        return p;
    };

    const size_t NX  = (size_t)MTOT * DMODEL;      // 8.4M
    const size_t NWQ = (size_t)DMODEL * NH * HD;   // 4.2M
    const size_t NWK = (size_t)DMODEL * NKV * HD;  // 1.05M
    __bf16* xb  = (__bf16*)alloc(NX * 2);
    __bf16* wqb = (__bf16*)alloc(NWQ * 2);
    __bf16* wkb = (__bf16*)alloc(NWK * 2);
    __bf16* wvb = (__bf16*)alloc(NWK * 2);
    __bf16* wob = (__bf16*)alloc(NWQ * 2);
    __bf16* qb  = (__bf16*)alloc((size_t)BATCH * NH * SEQ * HD * 2);
    __bf16* kb  = (__bf16*)alloc((size_t)BATCH * NKV * SEQ * HD * 2);
    __bf16* vb  = (__bf16*)alloc((size_t)BATCH * NKV * SEQ * HD * 2);
    __bf16* attnb = (__bf16*)alloc((size_t)BATCH * SEQ * NH * HD * 2);

    auto cvt = [&](const float* s, __bf16* d, size_t n) {
        int blocks = (int)((n / 4 + 255) / 256);
        cvt_f32_bf16<<<blocks, 256, 0, stream>>>(s, d, (int)n);
    };
    cvt(x, xb, NX);
    cvt(Wq, wqb, NWQ);
    cvt(Wk, wkb, NWK);
    cvt(Wv, wvb, NWK);
    cvt(Wo, wob, NWQ);

    qkv_kernel<<<dim3(MTOT / 64, NH + 2 * NKV), 256, 0, stream>>>(
        xb, wqb, wkb, wvb, qw, kw, qb, kb, vb);

    flash_kernel<<<dim3(SEQ / 64, NH, BATCH), 128, 0, stream>>>(qb, kb, vb, attnb);

    oproj_kernel<<<dim3(MTOT / 64, DMODEL / 128), 256, 0, stream>>>(attnb, wob, out);
}
